// QNetwork_41790031790731
// MI455X (gfx1250) — compile-verified, matrix operand order checked by reading, not run on hardware
//
#include <hip/hip_runtime.h>
#include <hip/hip_bf16.h>

typedef __attribute__((ext_vector_type(16))) _Float16 v16h;
typedef __attribute__((ext_vector_type(8)))  _Float16 v8h;
typedef __attribute__((ext_vector_type(8)))  float    v8f;

#define AA      8192      // B*N agents
#define SS      121       // 11*11
#define FW      11
#define HIDN    256
#define KNB     8
#define NPATCH  65536     // AA*KNB
#define PAIRS   32768     // B*P
#define MTILES  61952     // AA*SS/16

union HV { v16h v; v8h h[2]; };

__device__ __forceinline__ int kmap(int t, int h) { return h*8 + t + ((t >= 8) ? 8 : 0); }
__device__ __forceinline__ float lrelu(float v) { return v > 0.f ? v : 0.01f * v; }

__device__ __forceinline__ v8h zero8() {
  v8h z;
  for (int i = 0; i < 8; ++i) z[i] = (_Float16)0.f;
  return z;
}

__device__ __forceinline__ v8f wmma16(v16h a, v16h b, v8f c) {
  // D = A(16x32 f16) * B(32x16 f16) + C(f32)
  return __builtin_amdgcn_wmma_f32_16x16x32_f16(false, a, false, b, (short)0, c, false, false);
}

// ---------------- weight fragment prep ----------------
// FC weight W[256][K] (torch out,in) -> B fragments [kstep][ntile=16][lane=32][16 halfs]
__global__ void k_frag_fc(const float* __restrict__ W, _Float16* __restrict__ frag,
                          int K, int ksteps) {
  int idx = blockIdx.x * 256 + threadIdx.x;
  int total = ksteps * 16 * 512;
  if (idx >= total) return;
  int t    = idx & 15;
  int lane = (idx >> 4) & 31;
  int nt   = (idx >> 9) & 15;
  int ks   = idx >> 13;
  int h = lane >> 4;
  int k = ks * 32 + kmap(t, h);
  int n = nt * 16 + (lane & 15);
  float v = (k < K) ? W[(size_t)n * K + k] : 0.f;
  frag[idx] = (_Float16)v;
}

// Conv weight W[Cout][Cin][taps] -> per-tap B fragments [tp][ntile][lane][16]
__global__ void k_frag_conv(const float* __restrict__ W, _Float16* __restrict__ frag,
                            int Cin, int Cout, int ntiles, int taps) {
  int idx = blockIdx.x * 256 + threadIdx.x;
  int total = taps * ntiles * 512;
  if (idx >= total) return;
  int t    = idx & 15;
  int lane = (idx >> 4) & 31;
  int rest = idx >> 9;              // tp*ntiles + nt
  int nt = rest % ntiles;
  int tp = rest / ntiles;
  int h = lane >> 4;
  int c = kmap(t, h);
  int n = nt * 16 + (lane & 15);
  float v = (c < Cin && n < Cout) ? W[(size_t)(n * Cin + c) * taps + tp] : 0.f;
  frag[idx] = (_Float16)v;
}

// ---------------- input packing ----------------
// obs [A][8][11][11] f32 -> fov8 [A*121][8] f16 (ch7 zeroed), coords [A][2] f32
__global__ void k_pack_fov(const float* __restrict__ obs, _Float16* __restrict__ fov8,
                           float* __restrict__ coords) {
  int idx = blockIdx.x * 256 + threadIdx.x;
  if (idx >= AA * SS) return;
  int a = idx / SS;
  int pos = idx - a * SS;
  const float* op = obs + (size_t)a * 8 * SS;
  _Float16* dst = fov8 + (size_t)idx * 8;
#pragma unroll
  for (int c = 0; c < 8; ++c)
    dst[c] = (c < 7) ? (_Float16)op[c * SS + pos] : (_Float16)0.f;
  if (pos == 0) {
    coords[a * 2 + 0] = op[7 * SS + 0];
    coords[a * 2 + 1] = op[7 * SS + 1];
  }
}

// ---------------- implicit-GEMM conv via WMMA ----------------
// in: [A*121][CPIN] f16, out: [A*121][NTILES*16] f16, lrelu(+optional residual).
// One wave per 16-row M-tile; TAPS/NTILES/CPIN compile-time -> fully unrolled
// (TAPS*NTILES back-to-back v_wmma per tile).
template <int TAPS, int NTILES, int CPIN>
__global__ __launch_bounds__(256)
void k_conv_wmma(const _Float16* __restrict__ in,
                 const _Float16* __restrict__ frag,
                 const float* __restrict__ bias,
                 const _Float16* __restrict__ res,
                 _Float16* __restrict__ out) {
  constexpr int COUT = NTILES * 16;
  int wid  = threadIdx.x >> 5;
  int lane = threadIdx.x & 31;
  int tile = blockIdx.x * 8 + wid;
  if (tile >= MTILES) return;
  int h   = lane >> 4;
  int l15 = lane & 15;
  int g = tile * 16 + l15;          // global row for this lane's A-rows
  int a = g / SS;
  int pos = g - a * SS;
  int y = pos / FW;
  int x = pos - y * FW;
  int c0 = h * 8;

  v8f acc[NTILES];
#pragma unroll
  for (int i = 0; i < NTILES; ++i)
#pragma unroll
    for (int r = 0; r < 8; ++r) acc[i][r] = 0.f;

#pragma unroll
  for (int tp = 0; tp < TAPS; ++tp) {
    constexpr int off = (TAPS == 9) ? 1 : 0;
    const int dy = (TAPS == 9) ? (tp / 3) - off : 0;
    const int dx = (TAPS == 9) ? (tp % 3) - off : 0;
    int yy = y + dy, xx = x + dx;
    bool inb = (yy >= 0) && (yy < FW) && (xx >= 0) && (xx < FW);
    HV A;
    A.h[0] = zero8();
    A.h[1] = zero8();
    if (inb) {
      const _Float16* bp = in + ((size_t)a * SS + yy * FW + xx) * CPIN;
      if (c0 < CPIN)      A.h[0] = *(const v8h*)(bp + c0);
      if (c0 + 16 < CPIN) A.h[1] = *(const v8h*)(bp + c0 + 16);
    }
#pragma unroll
    for (int nt = 0; nt < NTILES; ++nt) {
      const _Float16* fp = frag + ((size_t)(tp * NTILES + nt) * 32 + lane) * 16;
      HV Bf;
      Bf.h[0] = *(const v8h*)fp;
      Bf.h[1] = *(const v8h*)(fp + 8);
      acc[nt] = wmma16(A.v, Bf.v, acc[nt]);
    }
  }
#pragma unroll
  for (int nt = 0; nt < NTILES; ++nt) {
    int col = nt * 16 + l15;
    float b = bias[col];
#pragma unroll
    for (int r = 0; r < 8; ++r) {
      int gg = tile * 16 + r + 8 * h;
      float v = acc[nt][r] + b;
      if (res) v += (float)res[(size_t)gg * COUT + col];
      out[(size_t)gg * COUT + col] = (_Float16)lrelu(v);
    }
  }
}

// ---------------- generic WMMA GEMM, N fixed = 256 ----------------
// MODE 0: X row-major [M][Ks]; MODE 1: conv-flatten remap (k = c*121+pos),
//   X stored [(row*121+pos)*Crm + c].
template <int MODE>
__global__ __launch_bounds__(256)
void k_gemm_wmma(const _Float16* __restrict__ X, int Ks, int Crm,
                 const _Float16* __restrict__ frag,
                 const float* __restrict__ bias, int act,
                 _Float16* __restrict__ out16, float* __restrict__ out32) {
  __shared__ __align__(16) _Float16 sA[512];   // 16 rows x 32 k
  int tid = threadIdx.x;
  int wid = tid >> 5, lane = tid & 31;
  int h = lane >> 4, l15 = lane & 15;
  int mtile = blockIdx.x;
  int ksteps = (Ks + 31) >> 5;

  v8f acc[2];
#pragma unroll
  for (int i = 0; i < 2; ++i)
#pragma unroll
    for (int r = 0; r < 8; ++r) acc[i][r] = 0.f;

  for (int ks = 0; ks < ksteps; ++ks) {
    for (int e = tid; e < 512; e += 256) {
      int r = e >> 5, kk = e & 31;
      int k = ks * 32 + kk;
      int row = mtile * 16 + r;
      _Float16 v = (_Float16)0.f;
      if (k < Ks) {
        if (MODE == 0) v = X[(size_t)row * Ks + k];
        else {
          int c = k / SS;
          int p = k - c * SS;
          v = X[((size_t)row * SS + p) * Crm + c];
        }
      }
      sA[e] = v;
    }
    __syncthreads();
    HV Af;
    Af.h[0] = *(const v8h*)&sA[l15 * 32 + h * 8];
    Af.h[1] = *(const v8h*)&sA[l15 * 32 + h * 8 + 16];
    if (ks + 1 < ksteps) {
      // prefetch next k-step's B fragments into cache (global_prefetch_b8)
      __builtin_prefetch(frag + ((size_t)((ks + 1) * 16 + wid * 2) * 32 + lane) * 16, 0, 1);
    }
#pragma unroll
    for (int i = 0; i < 2; ++i) {
      int nt = wid * 2 + i;
      const _Float16* fp = frag + ((size_t)(ks * 16 + nt) * 32 + lane) * 16;
      HV Bf;
      Bf.h[0] = *(const v8h*)fp;
      Bf.h[1] = *(const v8h*)(fp + 8);
      acc[i] = wmma16(Af.v, Bf.v, acc[i]);
    }
    __syncthreads();
  }
#pragma unroll
  for (int i = 0; i < 2; ++i) {
    int nt = wid * 2 + i;
    int col = nt * 16 + l15;
    float b = bias[col];
#pragma unroll
    for (int r = 0; r < 8; ++r) {
      int row = mtile * 16 + r + 8 * h;
      float v = acc[i][r] + b;
      if (act) v = lrelu(v);
      if (out16) out16[(size_t)row * 256 + col] = (_Float16)v;
      if (out32) out32[(size_t)row * 256 + col] = v;
    }
  }
}

// ---------------- concat [h_img | coord linear] -> xcat [A][320] f16 ----------------
__global__ void k_xcat(const _Float16* __restrict__ h_img, const float* __restrict__ coords,
                       const float* __restrict__ cw, const float* __restrict__ cb,
                       _Float16* __restrict__ xcat) {
  int idx = blockIdx.x * 256 + threadIdx.x;
  if (idx >= AA * 320) return;
  int a = idx / 320, c = idx - a * 320;
  _Float16 v;
  if (c < 256) v = h_img[(size_t)a * 256 + c];
  else {
    int j = c - 256;
    float t = cb[j] + coords[a * 2] * cw[j * 2] + coords[a * 2 + 1] * cw[j * 2 + 1];
    v = (_Float16)t;
  }
  xcat[idx] = v;
}

// ---------------- neighbor heuristic convs (tiny, VALU) ----------------
// patches [q][121] f32 -> nh2 [(q*121+pos)][4] f16 (two 3x3 convs, lrelu each)
__global__ void k_nh_conv(const float* __restrict__ patches,
                          const float* __restrict__ w1, const float* __restrict__ b1,
                          const float* __restrict__ w2, const float* __restrict__ b2,
                          _Float16* __restrict__ out) {
  int q = blockIdx.x;
  int tid = threadIdx.x;
  __shared__ float sp[121];
  __shared__ float sc1[121 * 8];
  if (tid < 121) sp[tid] = patches[(size_t)q * SS + tid];
  __syncthreads();
  if (tid < 121) {
    int y = tid / FW, x = tid - y * FW;
    for (int o = 0; o < 8; ++o) {
      float acc = b1[o];
      for (int dy = 0; dy < 3; ++dy)
        for (int dx = 0; dx < 3; ++dx) {
          int yy = y + dy - 1, xx = x + dx - 1;
          if (yy >= 0 && yy < FW && xx >= 0 && xx < FW)
            acc += w1[o * 9 + dy * 3 + dx] * sp[yy * FW + xx];
        }
      sc1[tid * 8 + o] = lrelu(acc);
    }
  }
  __syncthreads();
  if (tid < 121) {
    int y = tid / FW, x = tid - y * FW;
    for (int o = 0; o < 4; ++o) {
      float acc = b2[o];
      for (int dy = 0; dy < 3; ++dy)
        for (int dx = 0; dx < 3; ++dx) {
          int yy = y + dy - 1, xx = x + dx - 1;
          if (yy >= 0 && yy < FW && xx >= 0 && xx < FW) {
            const float* wp = w2 + (o * 8) * 9 + dy * 3 + dx;
            const float* cp = &sc1[(yy * FW + xx) * 8];
            for (int ci = 0; ci < 8; ++ci) acc += wp[ci * 9] * cp[ci];
          }
        }
      out[((size_t)q * SS + tid) * 4 + o] = (_Float16)lrelu(acc);
    }
  }
}

// ---------------- attention: scores + softmax ----------------
__global__ void k_scores(const _Float16* __restrict__ qb, const _Float16* __restrict__ kb,
                         float* __restrict__ attw) {
  int idx = blockIdx.x * 256 + threadIdx.x;
  if (idx >= AA * 2) return;
  int a = idx >> 1, hh = idx & 1;
  const _Float16* qp = qb + (size_t)a * 256 + hh * 128;
  float s[KNB];
  float mx = -1e30f;
  for (int j = 0; j < KNB; ++j) {
    const _Float16* kp = kb + ((size_t)(a * KNB + j)) * 256 + hh * 128;
    float d = 0.f;
    for (int t = 0; t < 128; ++t) d += (float)qp[t] * (float)kp[t];
    s[j] = d * 0.08838834764831845f;   // 1/sqrt(128)
    mx = fmaxf(mx, s[j]);
  }
  float sum = 0.f;
  for (int j = 0; j < KNB; ++j) { s[j] = __expf(s[j] - mx); sum += s[j]; }
  float inv = 1.f / sum;
  for (int j = 0; j < KNB; ++j) attw[(size_t)a * 16 + hh * 8 + j] = s[j] * inv;
}

__global__ void k_ctx(const float* __restrict__ attw, const _Float16* __restrict__ vb,
                      _Float16* __restrict__ ctx) {
  int idx = blockIdx.x * 256 + threadIdx.x;
  if (idx >= AA * 256) return;
  int a = idx >> 8, d = idx & 255, hh = d >> 7;
  float acc = 0.f;
  for (int j = 0; j < KNB; ++j)
    acc += attw[(size_t)a * 16 + hh * 8 + j] * (float)vb[((size_t)(a * KNB + j)) * 256 + d];
  ctx[idx] = (_Float16)acc;
}

// ---------------- pair gather: write pair_enc f32 output + f16 copy ----------------
__global__ void k_gather(const float* __restrict__ fused, const int* __restrict__ ia,
                         const int* __restrict__ ib, float* __restrict__ outp,
                         _Float16* __restrict__ pair16) {
  int idx = blockIdx.x * 256 + threadIdx.x;   // < PAIRS*512
  int pr = idx >> 9;
  int c  = idx & 511;
  int b  = pr >> 9;         // 512 pairs per batch
  int pp = pr & 511;
  int n = (c < 256) ? ia[b * 512 + pp] : ib[b * 512 + pp];
  float v = fused[((size_t)(b * 128 + n)) * 256 + (c & 255)];
  outp[idx] = v;
  pair16[idx] = (_Float16)v;
}

// ---------------- q4: [PAIRS][256] -> [PAIRS][3] ----------------
__global__ void k_q4(const _Float16* __restrict__ z, const float* __restrict__ W,
                     const float* __restrict__ b, float* __restrict__ outq) {
  int idx = blockIdx.x * 256 + threadIdx.x;
  if (idx >= PAIRS * 3) return;
  int pr = idx / 3, o = idx - pr * 3;
  float acc = b[o];
  const _Float16* zp = z + (size_t)pr * 256;
  for (int t = 0; t < 256; ++t) acc += (float)zp[t] * W[o * 256 + t];
  outq[idx] = acc;
}

extern "C" void kernel_launch(void* const* d_in, const int* in_sizes, int n_in,
                              void* d_out, int out_size, void* d_ws, size_t ws_size,
                              hipStream_t stream) {
  (void)in_sizes; (void)n_in; (void)out_size; (void)ws_size;
  const float* obs      = (const float*)d_in[0];
  const float* npat     = (const float*)d_in[1];
  const float* enc_c1_w = (const float*)d_in[2];  const float* enc_c1_b = (const float*)d_in[3];
  const float* enc_c2_w = (const float*)d_in[4];  const float* enc_c2_b = (const float*)d_in[5];
  const float* enc_fc_w = (const float*)d_in[6];  const float* enc_fc_b = (const float*)d_in[7];
  const float* coord_w  = (const float*)d_in[8];  const float* coord_b  = (const float*)d_in[9];
  const float* enc_out_w= (const float*)d_in[10]; const float* enc_out_b= (const float*)d_in[11];
  const float* nh_c1_w  = (const float*)d_in[12]; const float* nh_c1_b  = (const float*)d_in[13];
  const float* nh_c2_w  = (const float*)d_in[14]; const float* nh_c2_b  = (const float*)d_in[15];
  const float* nh_fc_w  = (const float*)d_in[16]; const float* nh_fc_b  = (const float*)d_in[17];
  const float* attn_wq  = (const float*)d_in[18]; const float* attn_bq  = (const float*)d_in[19];
  const float* attn_wk  = (const float*)d_in[20]; const float* attn_bk  = (const float*)d_in[21];
  const float* attn_wv  = (const float*)d_in[22]; const float* attn_bv  = (const float*)d_in[23];
  const float* attn_wo  = (const float*)d_in[24]; const float* attn_bo  = (const float*)d_in[25];
  const float* q1_w = (const float*)d_in[26]; const float* q1_b = (const float*)d_in[27];
  const float* q2_w = (const float*)d_in[28]; const float* q2_b = (const float*)d_in[29];
  const float* q3_w = (const float*)d_in[30]; const float* q3_b = (const float*)d_in[31];
  const float* q4_w = (const float*)d_in[32]; const float* q4_b = (const float*)d_in[33];
  const float* rbw[6]; const float* rbb[6];
  for (int i = 0; i < 6; ++i) { rbw[i] = (const float*)d_in[34 + 2*i]; rbb[i] = (const float*)d_in[35 + 2*i]; }
  const int* idx_a = (const int*)d_in[46];
  const int* idx_b = (const int*)d_in[47];

  // ---- workspace layout (bump allocator, 256B aligned) ----
  char* ws = (char*)d_ws;
  size_t off = 0;
  auto alloc = [&](size_t bytes) -> char* {
    off = (off + 255) & ~(size_t)255;
    char* p = ws + off;
    off += bytes;
    return p;
  };
  const size_t ACT32 = (size_t)AA * SS * 32 * 2;      // 63.4 MB
  _Float16* fov8   = (_Float16*)alloc((size_t)AA * SS * 8 * 2);
  float*    coords = (float*)   alloc((size_t)AA * 2 * 4);
  _Float16* bufX   = (_Float16*)alloc(ACT32);
  _Float16* bufY   = (_Float16*)alloc(ACT32);
  _Float16* bufZ   = (_Float16*)alloc(ACT32);
  _Float16* nh2    = (_Float16*)alloc(ACT32);         // also reused as kb
  _Float16* nb     = (_Float16*)alloc((size_t)NPATCH * 256 * 2);
  _Float16* vb     = (_Float16*)alloc((size_t)NPATCH * 256 * 2);
  _Float16* h_img  = (_Float16*)alloc((size_t)AA * 256 * 2);
  _Float16* xcat   = (_Float16*)alloc((size_t)AA * 320 * 2);
  _Float16* encf   = (_Float16*)alloc((size_t)AA * 256 * 2);
  _Float16* qb     = (_Float16*)alloc((size_t)AA * 256 * 2);
  float*    attw   = (float*)   alloc((size_t)AA * 16 * 4);
  _Float16* ctxb   = (_Float16*)alloc((size_t)AA * 256 * 2);
  float*    fused32= (float*)   alloc((size_t)AA * 256 * 4);
  _Float16* f_ec1  = (_Float16*)alloc(9 * 2 * 512 * 2);
  _Float16* f_rb[6];
  for (int i = 0; i < 6; ++i) f_rb[i] = (_Float16*)alloc(9 * 2 * 512 * 2);
  _Float16* f_ec2  = (_Float16*)alloc(1 * 1 * 512 * 2);
  _Float16* f_efc  = (_Float16*)alloc((size_t)61 * 16 * 512 * 2);
  _Float16* f_nfc  = (_Float16*)alloc((size_t)16 * 16 * 512 * 2);
  _Float16* f_eo   = (_Float16*)alloc((size_t)10 * 16 * 512 * 2);
  _Float16* f_wq   = (_Float16*)alloc((size_t)8 * 16 * 512 * 2);
  _Float16* f_wk   = (_Float16*)alloc((size_t)8 * 16 * 512 * 2);
  _Float16* f_wv   = (_Float16*)alloc((size_t)8 * 16 * 512 * 2);
  _Float16* f_wo   = (_Float16*)alloc((size_t)8 * 16 * 512 * 2);
  _Float16* f_q1   = (_Float16*)alloc((size_t)16 * 16 * 512 * 2);
  _Float16* f_q2   = (_Float16*)alloc((size_t)8 * 16 * 512 * 2);
  _Float16* f_q3   = (_Float16*)alloc((size_t)8 * 16 * 512 * 2);
  // lifetime-based aliases:
  _Float16* enc16  = bufY;                  // enc_c2 output [A*121][16] (bufY free then)
  _Float16* kb     = nh2;                   // K projection (nh2 consumed by nh_fc)
  _Float16* pair16 = bufZ;                  // pair features (bufZ free after rb2)
  _Float16* z1     = bufX;                  // q1 out
  _Float16* z2     = bufX + (size_t)PAIRS * 256;  // q2 out (still inside bufX)
  _Float16* z3     = bufY;                  // q3 out (bufY free after enc_fc)

  // ---- 1. weight fragment prep ----
  k_frag_fc<<<1952, 256, 0, stream>>>(enc_fc_w, f_efc, 1936, 61);
  k_frag_fc<<< 512, 256, 0, stream>>>(nh_fc_w,  f_nfc,  484, 16);
  k_frag_fc<<< 320, 256, 0, stream>>>(enc_out_w,f_eo,   320, 10);
  k_frag_fc<<< 256, 256, 0, stream>>>(attn_wq,  f_wq,   256,  8);
  k_frag_fc<<< 256, 256, 0, stream>>>(attn_wk,  f_wk,   256,  8);
  k_frag_fc<<< 256, 256, 0, stream>>>(attn_wv,  f_wv,   256,  8);
  k_frag_fc<<< 256, 256, 0, stream>>>(attn_wo,  f_wo,   256,  8);
  k_frag_fc<<< 512, 256, 0, stream>>>(q1_w,     f_q1,   512, 16);
  k_frag_fc<<< 256, 256, 0, stream>>>(q2_w,     f_q2,   256,  8);
  k_frag_fc<<< 256, 256, 0, stream>>>(q3_w,     f_q3,   256,  8);
  k_frag_conv<<<36, 256, 0, stream>>>(enc_c1_w, f_ec1, 7, 32, 2, 9);
  for (int i = 0; i < 6; ++i)
    k_frag_conv<<<36, 256, 0, stream>>>(rbw[i], f_rb[i], 32, 32, 2, 9);
  k_frag_conv<<<2, 256, 0, stream>>>(enc_c2_w, f_ec2, 32, 16, 1, 1);

  // ---- 2. pack obs ----
  k_pack_fov<<<3872, 256, 0, stream>>>(obs, fov8, coords);

  // ---- 3. encoder conv stack (WMMA implicit GEMM, fully unrolled taps) ----
  k_conv_wmma<9,2, 8><<<7744, 256, 0, stream>>>(fov8, f_ec1,   enc_c1_b, nullptr, bufX);
  k_conv_wmma<9,2,32><<<7744, 256, 0, stream>>>(bufX, f_rb[0], rbb[0],   nullptr, bufY);
  k_conv_wmma<9,2,32><<<7744, 256, 0, stream>>>(bufY, f_rb[1], rbb[1],   bufX,    bufZ);
  k_conv_wmma<9,2,32><<<7744, 256, 0, stream>>>(bufZ, f_rb[2], rbb[2],   nullptr, bufX);
  k_conv_wmma<9,2,32><<<7744, 256, 0, stream>>>(bufX, f_rb[3], rbb[3],   bufZ,    bufY);
  k_conv_wmma<9,2,32><<<7744, 256, 0, stream>>>(bufY, f_rb[4], rbb[4],   nullptr, bufZ);
  k_conv_wmma<9,2,32><<<7744, 256, 0, stream>>>(bufZ, f_rb[5], rbb[5],   bufY,    bufX);
  k_conv_wmma<1,1,32><<<7744, 256, 0, stream>>>(bufX, f_ec2,   enc_c2_b, nullptr, enc16);

  // ---- 4. enc FC path ----
  k_gemm_wmma<1><<<512, 256, 0, stream>>>(enc16, 1936, 16, f_efc, enc_fc_b, 1, h_img, nullptr);
  k_xcat<<<10240, 256, 0, stream>>>(h_img, coords, coord_w, coord_b, xcat);
  k_gemm_wmma<0><<<512, 256, 0, stream>>>(xcat, 320, 0, f_eo, enc_out_b, 0, encf, nullptr);

  // ---- 5. neighbor path ----
  k_nh_conv<<<NPATCH, 128, 0, stream>>>(npat, nh_c1_w, nh_c1_b, nh_c2_w, nh_c2_b, nh2);
  k_gemm_wmma<1><<<4096, 256, 0, stream>>>(nh2, 484, 4, f_nfc, nh_fc_b, 1, nb, nullptr);

  // ---- 6. attention ----
  k_gemm_wmma<0><<< 512, 256, 0, stream>>>(encf, 256, 0, f_wq, attn_bq, 0, qb, nullptr);
  k_gemm_wmma<0><<<4096, 256, 0, stream>>>(nb,   256, 0, f_wk, attn_bk, 0, kb, nullptr);
  k_gemm_wmma<0><<<4096, 256, 0, stream>>>(nb,   256, 0, f_wv, attn_bv, 0, vb, nullptr);
  k_scores<<<64, 256, 0, stream>>>(qb, kb, attw);
  k_ctx<<<8192, 256, 0, stream>>>(attw, vb, ctxb);
  k_gemm_wmma<0><<<512, 256, 0, stream>>>(ctxb, 256, 0, f_wo, attn_bo, 0, nullptr, fused32);

  // ---- 7. pair gather + qnet ----
  float* out_pair = (float*)d_out;
  float* out_q    = out_pair + (size_t)PAIRS * 512;
  k_gather<<<65536, 256, 0, stream>>>(fused32, idx_a, idx_b, out_pair, pair16);
  k_gemm_wmma<0><<<2048, 256, 0, stream>>>(pair16, 512, 0, f_q1, q1_b, 1, z1, nullptr);
  k_gemm_wmma<0><<<2048, 256, 0, stream>>>(z1,     256, 0, f_q2, q2_b, 1, z2, nullptr);
  k_gemm_wmma<0><<<2048, 256, 0, stream>>>(z2,     256, 0, f_q3, q3_b, 1, z3, nullptr);
  k_q4<<<384, 256, 0, stream>>>(z3, q4_w, q4_b, out_q);
}
